// SpatBranch_89601607729228
// MI455X (gfx1250) — compile-verified, matrix operand order checked by reading, not run on hardware
//
#include <hip/hip_runtime.h>
#include <hip/hip_fp16.h>

typedef __attribute__((ext_vector_type(16))) _Float16 v16h;
typedef __attribute__((ext_vector_type(8)))  _Float16 v8h;
typedef __attribute__((ext_vector_type(8)))  float    v8f;

#define NN    50000
#define EE    400000
#define MM    (EE + NN)     // edges + self loops
#define DIN   128
#define F1    512           // H1*C1
#define H1N   8
#define C1N   64
#define F2    16
#define SLOPE 0.2f

static __device__ __forceinline__ float lrelu(float x) { return x > 0.f ? x : SLOPE * x; }
static __device__ __forceinline__ float eluf(float x)  { return x > 0.f ? x : __expf(x) - 1.f; }

// ---------------- CSR construction ----------------
__global__ void k_zero_counts(int* counts) {
    int i = blockIdx.x * blockDim.x + threadIdx.x;
    if (i < NN) counts[i] = 0;
}

__global__ void k_count(const int* __restrict__ ei, int* counts) {
    int e = blockIdx.x * blockDim.x + threadIdx.x;
    if (e >= MM) return;
    int dst = (e < EE) ? ei[EE + e] : (e - EE);   // self loop dst = node id
    atomicAdd(&counts[dst], 1);
}

__global__ void k_scan(const int* __restrict__ counts, int* row_ptr, int* write_ptr) {
    __shared__ int sums[1024];
    const int CH = (NN + 1023) / 1024;
    int t = threadIdx.x;
    int beg = t * CH;
    int end = beg + CH; if (end > NN) end = NN; if (beg > NN) beg = NN;
    int s = 0;
    for (int i = beg; i < end; ++i) s += counts[i];
    sums[t] = s;
    __syncthreads();
    for (int off = 1; off < 1024; off <<= 1) {
        int v = (t >= off) ? sums[t - off] : 0;
        __syncthreads();
        sums[t] += v;
        __syncthreads();
    }
    int run = (t == 0) ? 0 : sums[t - 1];
    for (int i = beg; i < end; ++i) {
        row_ptr[i] = run; write_ptr[i] = run;
        run += counts[i];
    }
    if (t == 1023) row_ptr[NN] = sums[1023];
}

__global__ void k_scatter(const int* __restrict__ ei, int* write_ptr, int* col) {
    int e = blockIdx.x * blockDim.x + threadIdx.x;
    if (e >= MM) return;
    int src, dst;
    if (e < EE) { src = ei[e]; dst = ei[EE + e]; }
    else        { src = e - EE; dst = e - EE; }
    int pos = atomicAdd(&write_ptr[dst], 1);
    col[pos] = src;
}

// ------------- prep: transpose + f16-convert weights for WMMA B operands ----
// W1T[c*128 + k] = W1[k*512 + c]   (512 x 128, K contiguous)
// W2T[c*512 + k] = W2[k*16  + c]   (16  x 512, K contiguous)
__global__ void k_prep_w(const float* __restrict__ W1, const float* __restrict__ W2,
                         _Float16* W1T, _Float16* W2T) {
    int i = blockIdx.x * blockDim.x + threadIdx.x;
    if (i < DIN * F1) {
        int c = i >> 7, k = i & 127;
        W1T[i] = (_Float16)W1[k * F1 + c];
    } else {
        int j = i - DIN * F1;
        if (j < F1 * F2) {
            int c = j >> 9, k = j & 511;
            W2T[j] = (_Float16)W2[k * F2 + c];
        }
    }
}

// ---------------- GEMM1: h = x @ W1 ; 16x64 strip per wave, 4 WMMA accs -----
__global__ void __launch_bounds__(256) k_gemm1(const float* __restrict__ x,
                                               const _Float16* __restrict__ W1T,
                                               _Float16* __restrict__ hfeat) {
    int wid  = (blockIdx.x * blockDim.x + threadIdx.x) >> 5;
    int lane = threadIdx.x & 31;
    int tm = wid >> 3;          // 3125 row tiles
    int tq = wid & 7;           // 8 groups of 64 columns
    if (tm >= NN / 16) return;

    int row     = tm * 16 + (lane & 15);
    int khalf   = (lane >> 4) * 8;           // K sub-half per lane group
    int colbase = tq * 64 + (lane & 15);

    v8f acc0 = {}, acc1 = {}, acc2 = {}, acc3 = {};
    for (int ks = 0; ks < DIN; ks += 32) {
        int kb = ks + khalf;
        const float* xr = x + (size_t)row * DIN + kb;
        v16h a;
#pragma unroll
        for (int i = 0; i < 8; ++i) {
            a[i]     = (_Float16)xr[i];        // k = kb .. kb+7
            a[i + 8] = (_Float16)xr[i + 16];   // k = kb+16 .. kb+23
        }
        v16h b0, b1, b2, b3;
        {
            const v8h* w0 = (const v8h*)(W1T + (size_t)(colbase)      * DIN + kb);
            const v8h* w1 = (const v8h*)(W1T + (size_t)(colbase + 16) * DIN + kb);
            const v8h* w2 = (const v8h*)(W1T + (size_t)(colbase + 32) * DIN + kb);
            const v8h* w3 = (const v8h*)(W1T + (size_t)(colbase + 48) * DIN + kb);
            v8h l0 = w0[0], h0 = w0[2];
            v8h l1 = w1[0], h1 = w1[2];
            v8h l2 = w2[0], h2 = w2[2];
            v8h l3 = w3[0], h3 = w3[2];
#pragma unroll
            for (int i = 0; i < 8; ++i) {
                b0[i] = l0[i]; b0[i + 8] = h0[i];
                b1[i] = l1[i]; b1[i + 8] = h1[i];
                b2[i] = l2[i]; b2[i + 8] = h2[i];
                b3[i] = l3[i]; b3[i + 8] = h3[i];
            }
        }
        acc0 = __builtin_amdgcn_wmma_f32_16x16x32_f16(false, a, false, b0, (short)0, acc0, false, false);
        acc1 = __builtin_amdgcn_wmma_f32_16x16x32_f16(false, a, false, b1, (short)0, acc1, false, false);
        acc2 = __builtin_amdgcn_wmma_f32_16x16x32_f16(false, a, false, b2, (short)0, acc2, false, false);
        acc3 = __builtin_amdgcn_wmma_f32_16x16x32_f16(false, a, false, b3, (short)0, acc3, false, false);
    }
    int mbase = tm * 16 + (lane >> 4) * 8;
#pragma unroll
    for (int r = 0; r < 8; ++r) {
        int off = (mbase + r) * F1;
        hfeat[off + colbase]      = (_Float16)acc0[r];
        hfeat[off + colbase + 16] = (_Float16)acc1[r];
        hfeat[off + colbase + 32] = (_Float16)acc2[r];
        hfeat[off + colbase + 48] = (_Float16)acc3[r];
    }
}

// ---------------- per-node attention halves, layer 1 ----------------
__global__ void k_attn1(const _Float16* __restrict__ hfeat,
                        const float* __restrict__ a_src, const float* __restrict__ a_dst,
                        float* al_src, float* al_dst) {
    int idx = blockIdx.x * blockDim.x + threadIdx.x;   // n*8 + h
    if (idx >= NN * H1N) return;
    int n = idx >> 3, h = idx & 7;
    float s = 0.f, d = 0.f;
#pragma unroll 8
    for (int c = 0; c < C1N; ++c) {
        float v = (float)hfeat[n * F1 + h * C1N + c];
        s += v * a_src[h * C1N + c];
        d += v * a_dst[h * C1N + c];
    }
    al_src[idx] = s; al_dst[idx] = d;
}

// ---------------- layer-1 softmax + aggregate (one block per dst node) -------
__global__ void __launch_bounds__(256) k_agg1(const int* __restrict__ row_ptr,
                                              const int* __restrict__ col,
                                              const _Float16* __restrict__ hfeat,
                                              const float* __restrict__ al_src,
                                              const float* __restrict__ al_dst,
                                              const float* __restrict__ b1,
                                              _Float16* __restrict__ h1act) {
    int node = blockIdx.x;
    int t = threadIdx.x;
    int head = t >> 5;          // wave id = head
    int lane = t & 31;
    __shared__ float s_m[H1N], s_d[H1N];

    int beg = row_ptr[node], end = row_ptr[node + 1];
    float ad = al_dst[node * H1N + head];

    float mx = -__builtin_inff();
    for (int i = beg + lane; i < end; i += 32)
        mx = fmaxf(mx, lrelu(al_src[col[i] * H1N + head] + ad));
    for (int o = 16; o; o >>= 1) mx = fmaxf(mx, __shfl_xor(mx, o, 32));
    float sm = 0.f;
    for (int i = beg + lane; i < end; i += 32)
        sm += __expf(lrelu(al_src[col[i] * H1N + head] + ad) - mx);
    for (int o = 16; o; o >>= 1) sm += __shfl_xor(sm, o, 32);
    if (lane == 0) { s_m[head] = mx; s_d[head] = sm; }
    __syncthreads();

    float m = s_m[head];
    float inv = 1.0f / s_d[head];   // >= 1 edge (self loop) always
    float acc0 = 0.f, acc1 = 0.f;
    for (int i = beg; i < end; ++i) {
        int s = col[i];
        if (i + 1 < end)   // warm L0/L2 for the next gathered feature row
            __builtin_prefetch(hfeat + (size_t)col[i + 1] * F1 + head * C1N, 0, 0);
        float alpha = __expf(lrelu(al_src[s * H1N + head] + ad) - m) * inv;
        const _Float16* hp = hfeat + (size_t)s * F1 + head * C1N;
        acc0 += alpha * (float)hp[lane];
        acc1 += alpha * (float)hp[lane + 32];
    }
    int o = node * F1 + head * C1N + lane;
    h1act[o]      = (_Float16)eluf(acc0 + b1[head * C1N + lane]);
    h1act[o + 32] = (_Float16)eluf(acc1 + b1[head * C1N + lane + 32]);
}

// ---------------- GEMM2: h2 = h1act @ W2  (all-f16 operands, WMMA) ----------
__global__ void __launch_bounds__(256) k_gemm2(const _Float16* __restrict__ h1act,
                                               const _Float16* __restrict__ W2T,
                                               float* __restrict__ h2) {
    int wid  = (blockIdx.x * blockDim.x + threadIdx.x) >> 5;
    int lane = threadIdx.x & 31;
    int tm = wid;                       // single col tile (F2 = 16)
    if (tm >= NN / 16) return;

    int row   = tm * 16 + (lane & 15);
    int colb  = lane & 15;
    int khalf = (lane >> 4) * 8;

    v8f acc = {};
    for (int ks = 0; ks < F1; ks += 32) {
        int kb = ks + khalf;
        const v8h* ar = (const v8h*)(h1act + (size_t)row * F1 + kb);
        const v8h* wr = (const v8h*)(W2T  + (size_t)colb * F1 + kb);
        v8h al = ar[0], ah = ar[2];
        v8h bl = wr[0], bh = wr[2];
        v16h a, b;
#pragma unroll
        for (int i = 0; i < 8; ++i) {
            a[i] = al[i]; a[i + 8] = ah[i];
            b[i] = bl[i]; b[i + 8] = bh[i];
        }
        acc = __builtin_amdgcn_wmma_f32_16x16x32_f16(false, a, false, b,
                                                     (short)0, acc, false, false);
    }
    int mbase = tm * 16 + (lane >> 4) * 8;
#pragma unroll
    for (int r = 0; r < 8; ++r)
        h2[(mbase + r) * F2 + colb] = acc[r];
}

// ---------------- per-node attention halves, layer 2 ----------------
__global__ void k_attn2(const float* __restrict__ h2,
                        const float* __restrict__ a_src, const float* __restrict__ a_dst,
                        float* al_src, float* al_dst) {
    int n = blockIdx.x * blockDim.x + threadIdx.x;
    if (n >= NN) return;
    float s = 0.f, d = 0.f;
#pragma unroll
    for (int c = 0; c < F2; ++c) {
        float v = h2[n * F2 + c];
        s += v * a_src[c];
        d += v * a_dst[c];
    }
    al_src[n] = s; al_dst[n] = d;
}

// ---------------- layer-2 softmax + aggregate (one wave per dst node) -------
__global__ void __launch_bounds__(256) k_agg2(const int* __restrict__ row_ptr,
                                              const int* __restrict__ col,
                                              const float* __restrict__ h2,
                                              const float* __restrict__ al_src,
                                              const float* __restrict__ al_dst,
                                              const float* __restrict__ b2,
                                              float* __restrict__ out) {
    int node = blockIdx.x * 8 + (threadIdx.x >> 5);
    if (node >= NN) return;
    int lane = threadIdx.x & 31;
    int beg = row_ptr[node], end = row_ptr[node + 1];
    float ad = al_dst[node];

    float mx = -__builtin_inff();
    for (int i = beg + lane; i < end; i += 32)
        mx = fmaxf(mx, lrelu(al_src[col[i]] + ad));
    for (int o = 16; o; o >>= 1) mx = fmaxf(mx, __shfl_xor(mx, o, 32));
    float sm = 0.f;
    for (int i = beg + lane; i < end; i += 32)
        sm += __expf(lrelu(al_src[col[i]] + ad) - mx);
    for (int o = 16; o; o >>= 1) sm += __shfl_xor(sm, o, 32);
    float inv = 1.0f / sm;

    if (lane < F2) {
        float acc = 0.f;
        for (int i = beg; i < end; ++i) {
            int s = col[i];
            float alpha = __expf(lrelu(al_src[s] + ad) - mx) * inv;
            acc += alpha * h2[s * F2 + lane];
        }
        out[node * F2 + lane] = acc + b2[lane];
    }
}

// ---------------- launcher ----------------
extern "C" void kernel_launch(void* const* d_in, const int* in_sizes, int n_in,
                              void* d_out, int out_size, void* d_ws, size_t ws_size,
                              hipStream_t stream) {
    const float* x     = (const float*)d_in[0];
    const int*   ei    = (const int*)  d_in[1];
    const float* W1    = (const float*)d_in[2];
    const float* a1s   = (const float*)d_in[3];
    const float* a1d   = (const float*)d_in[4];
    const float* b1    = (const float*)d_in[5];
    const float* W2    = (const float*)d_in[6];
    const float* a2s   = (const float*)d_in[7];
    const float* a2d   = (const float*)d_in[8];
    const float* b2    = (const float*)d_in[9];
    float* out = (float*)d_out;

    char* p = (char*)d_ws;
    auto carve = [&](size_t bytes) {
        void* r = (void*)p;
        p += (bytes + 255) & ~(size_t)255;
        return r;
    };
    int*       counts    = (int*)      carve(NN * sizeof(int));
    int*       row_ptr   = (int*)      carve((NN + 1) * sizeof(int));
    int*       write_ptr = (int*)      carve(NN * sizeof(int));
    int*       col       = (int*)      carve(MM * sizeof(int));
    _Float16*  hfeat     = (_Float16*) carve((size_t)NN * F1 * sizeof(_Float16));
    _Float16*  h1act     = (_Float16*) carve((size_t)NN * F1 * sizeof(_Float16));
    _Float16*  W1T       = (_Float16*) carve((size_t)DIN * F1 * sizeof(_Float16));
    _Float16*  W2T       = (_Float16*) carve((size_t)F1 * F2 * sizeof(_Float16));
    float*     al1s      = (float*)    carve((size_t)NN * H1N * sizeof(float));
    float*     al1d      = (float*)    carve((size_t)NN * H1N * sizeof(float));
    float*     h2        = (float*)    carve((size_t)NN * F2 * sizeof(float));
    float*     al2s      = (float*)    carve(NN * sizeof(float));
    float*     al2d      = (float*)    carve(NN * sizeof(float));

    // CSR build
    k_zero_counts<<<(NN + 255) / 256, 256, 0, stream>>>(counts);
    k_count<<<(MM + 255) / 256, 256, 0, stream>>>(ei, counts);
    k_scan<<<1, 1024, 0, stream>>>(counts, row_ptr, write_ptr);
    k_scatter<<<(MM + 255) / 256, 256, 0, stream>>>(ei, write_ptr, col);

    // Weight prep (transpose + f16)
    k_prep_w<<<(DIN * F1 + F1 * F2 + 255) / 256, 256, 0, stream>>>(W1, W2, W1T, W2T);

    // Layer 1
    k_gemm1<<<(NN / 16) * 8 / 8, 256, 0, stream>>>(x, W1T, hfeat);
    k_attn1<<<(NN * H1N + 255) / 256, 256, 0, stream>>>(hfeat, a1s, a1d, al1s, al1d);
    k_agg1<<<NN, 256, 0, stream>>>(row_ptr, col, hfeat, al1s, al1d, b1, h1act);

    // Layer 2
    k_gemm2<<<((NN / 16) + 7) / 8, 256, 0, stream>>>(h1act, W2T, h2);
    k_attn2<<<(NN + 255) / 256, 256, 0, stream>>>(h2, a2s, a2d, al2s, al2d);
    k_agg2<<<(NN + 7) / 8, 256, 0, stream>>>(row_ptr, col, h2, al2s, al2d, b2, out);
}
